// SelfAttentionHead_37881611551359
// MI455X (gfx1250) — matrix operand order repaired and checked by reading, not compiled, not run on hardware
//
#include <hip/hip_runtime.h>
#include <hip/hip_bf16.h>

// ---------------------------------------------------------------------------
// Single-head causal self-attention, MI455X (gfx1250, wave32, WMMA bf16)
//   B=16, T=2048, DM=1024, DK=128
// Kernel 1a: Q/K projection (row-major bf16), Kernel 1b: V^T projection.
// Kernel 2 : causal flash-attention fwd with v_wmma_f32_16x16x32_bf16.
// All epilogues restage through LDS for vectorized b128 global stores.
// Workspace: 3 * B*T*DK bf16 = 24 MB (fits in 192 MB L2)
// ---------------------------------------------------------------------------

typedef __attribute__((ext_vector_type(16))) __bf16 v16bf;
typedef __attribute__((ext_vector_type(8)))  __bf16 v8bf;
typedef __attribute__((ext_vector_type(8)))  float  v8f;

union V16 { v16bf v; v8bf h[2]; };

#define B_   16
#define T_   2048
#define DM_  1024
#define DK_  128

#define LS   40    // X/W LDS row stride (bf16): 80 B = 5*16 B
#define CS   136   // C LDS row stride (bf16): 272 B = 17*16 B
#define OS   132   // O LDS row stride (f32):  528 B = 33*16 B

static __device__ __forceinline__ v8f wmma_bf16(const V16& a, const V16& b, v8f c) {
  // D = A(16x32 bf16) * B(32x16 bf16) + C(16x16 f32)
  return __builtin_amdgcn_wmma_f32_16x16x32_bf16(false, a.v, false, b.v,
                                                 (short)0, c, false, false);
}

// ---------------------------------------------------------------------------
// Shared GEMM body: computes a 128x128 tile of X@W (K=1024) in bf16 and
// leaves it staged in ct[128][CS]. 256 threads = 8 waves in a 2(M) x 4(N)
// grid; each wave owns 64x32 = 4x2 WMMA fragments.
// Ends with a __syncthreads() so callers can read any part of ct.
// ---------------------------------------------------------------------------
static __device__ __forceinline__ void gemm_tile_to_lds(
    const float* __restrict__ x, const float* __restrict__ W, int m0,
    __bf16 (*xs)[LS], __bf16 (*wt)[LS], __bf16 (*ct)[CS])
{
  const int tid  = threadIdx.x;
  const int lane = tid & 31;
  const int wave = tid >> 5;
  const int wm   = wave >> 2;               // 0..1 -> 64 rows each
  const int wn   = wave & 3;                // 0..3 -> 32 cols each
  const int lm   = lane & 15;
  const int hi   = lane >> 4;

  v8f acc[4][2];
  #pragma unroll
  for (int i = 0; i < 4; ++i)
    #pragma unroll
    for (int j = 0; j < 2; ++j) acc[i][j] = (v8f){};

  for (int k0 = 0; k0 < DM_; k0 += 32) {
    // --- stage X tile (fp32 -> bf16): 1024 float4 over 256 threads ---
    #pragma unroll
    for (int i = 0; i < 4; ++i) {
      const int u   = tid + i * 256;        // 0..1023
      const int row = u >> 3;               // 0..127
      const int c   = (u & 7) * 4;          // 0..28
      const float4 f = *(const float4*)(x + (size_t)(m0 + row) * DM_ + k0 + c);
      xs[row][c + 0] = (__bf16)f.x;
      xs[row][c + 1] = (__bf16)f.y;
      xs[row][c + 2] = (__bf16)f.z;
      xs[row][c + 3] = (__bf16)f.w;
    }
    // --- stage W tile transposed (fp32 -> bf16) ---
    #pragma unroll
    for (int i = 0; i < 4; ++i) {
      const int u   = tid + i * 256;
      const int row = u >> 5;               // k-local 0..31
      const int c   = (u & 31) * 4;         // n 0..124
      const float4 f = *(const float4*)(W + (size_t)(k0 + row) * DK_ + c);
      wt[c + 0][row] = (__bf16)f.x;
      wt[c + 1][row] = (__bf16)f.y;
      wt[c + 2][row] = (__bf16)f.z;
      wt[c + 3][row] = (__bf16)f.w;
    }
    __syncthreads();

    // --- fragments + 8 WMMAs for this 32-deep K slab ---
    V16 a[4], b[2];
    #pragma unroll
    for (int mf = 0; mf < 4; ++mf) {
      const __bf16* base = &xs[wm * 64 + mf * 16 + lm][0];
      a[mf].h[0] = *(const v8bf*)(base + hi * 8);
      a[mf].h[1] = *(const v8bf*)(base + 16 + hi * 8);
    }
    #pragma unroll
    for (int nf = 0; nf < 2; ++nf) {
      const __bf16* base = &wt[wn * 32 + nf * 16 + lm][0];
      b[nf].h[0] = *(const v8bf*)(base + hi * 8);
      b[nf].h[1] = *(const v8bf*)(base + 16 + hi * 8);
    }
    #pragma unroll
    for (int mf = 0; mf < 4; ++mf)
      #pragma unroll
      for (int nf = 0; nf < 2; ++nf)
        acc[mf][nf] = wmma_bf16(a[mf], b[nf], acc[mf][nf]);

    __syncthreads();
  }

  // --- stage C tile (bf16) into LDS: C-layout scatter, then barrier ---
  #pragma unroll
  for (int mf = 0; mf < 4; ++mf)
    #pragma unroll
    for (int nf = 0; nf < 2; ++nf)
      #pragma unroll
      for (int r = 0; r < 8; ++r)
        ct[wm * 64 + mf * 16 + r + hi * 8][wn * 32 + nf * 16 + lm] =
            (__bf16)acc[mf][nf][r];
  __syncthreads();
}

// ---------------------------------------------------------------------------
// Kernel 1a: row-major projection. blockIdx.y: 0 -> (Wq,qws), 1 -> (Wk,kws).
// ---------------------------------------------------------------------------
__global__ __launch_bounds__(256, 2) void proj_rowmajor_kernel(
    const float* __restrict__ x,
    const float* __restrict__ Wq, const float* __restrict__ Wk,
    __bf16* __restrict__ qws, __bf16* __restrict__ kws)
{
  __shared__ __bf16 xs[128][LS];
  __shared__ __bf16 wt[128][LS];
  __shared__ __bf16 ct[128][CS];

  const float*  W   = blockIdx.y ? Wk  : Wq;
  __bf16*       dst = blockIdx.y ? kws : qws;
  const int m0 = blockIdx.x * 128;

  gemm_tile_to_lds(x, W, m0, xs, wt, ct);

  // vectorized store: 2048 chunks of 8 bf16 (16 B) over 256 threads
  const int tid = threadIdx.x;
  #pragma unroll
  for (int i = 0; i < 8; ++i) {
    const int u   = tid + i * 256;          // 0..2047
    const int row = u >> 4;                 // 0..127
    const int c8  = (u & 15) * 8;           // 0..120
    *(v8bf*)(dst + (size_t)(m0 + row) * DK_ + c8) = *(const v8bf*)&ct[row][c8];
  }
}

// ---------------------------------------------------------------------------
// Kernel 1b: transposed projection V^T [B][DK][T].
// ---------------------------------------------------------------------------
__global__ __launch_bounds__(256, 2) void proj_transpose_kernel(
    const float* __restrict__ x,
    const float* __restrict__ Wv,
    __bf16* __restrict__ vtws)
{
  __shared__ __bf16 xs[128][LS];
  __shared__ __bf16 wt[128][LS];
  __shared__ __bf16 ct[128][CS];

  const int m0 = blockIdx.x * 128;
  const int bb = m0 >> 11;                  // m0 / T_ (tile never crosses batch)
  const int t0 = m0 & (T_ - 1);

  gemm_tile_to_lds(x, Wv, m0, xs, wt, ct);

  // transposed vectorized store: gather a column strip from LDS, emit 16 B
  const int tid = threadIdx.x;
  #pragma unroll
  for (int i = 0; i < 8; ++i) {
    const int u   = tid + i * 256;          // 0..2047
    const int col = u >> 4;                 // dk index 0..127
    const int r8  = (u & 15) * 8;           // row-strip base 0..120
    v8bf val;
    #pragma unroll
    for (int j = 0; j < 8; ++j) val[j] = ct[r8 + j][col];
    *(v8bf*)(vtws + ((size_t)bb * DK_ + col) * T_ + t0 + r8) = val;
  }
}

// ---------------------------------------------------------------------------
// Kernel 2: causal flash attention forward.
// grid (T/64, B); 128 threads = 4 waves; wave owns 16 q rows.
// Per KV block of 64: S = Q K^T (16 WMMA), online softmax, O += P V (16 WMMA).
// ---------------------------------------------------------------------------
__global__ __launch_bounds__(128, 2) void attn_fwd_kernel(
    const __bf16* __restrict__ qws,   // [B*T, DK]
    const __bf16* __restrict__ kws,   // [B*T, DK]
    const __bf16* __restrict__ vtws,  // [B][DK][T]
    float* __restrict__ out)          // [B*T, DK]
{
  constexpr int PS = 72;                       // P row stride, 144 B = 9*16 B
  __shared__ __bf16 p_lds[4][16][PS];          // wave-private P staging
  __shared__ float  o_lds[4][16][OS];          // wave-private O staging

  const int bb    = blockIdx.y;
  const int q0    = blockIdx.x * 64;
  const int tid   = threadIdx.x;
  const int lane  = tid & 31;
  const int wave  = tid >> 5;
  const int lm    = lane & 15;
  const int hi    = lane >> 4;
  const int qrow0 = q0 + wave * 16;

  const __bf16* __restrict__ qb = qws  + (size_t)bb * T_ * DK_;
  const __bf16* __restrict__ kb = kws  + (size_t)bb * T_ * DK_;
  const __bf16* __restrict__ vb = vtws + (size_t)bb * DK_ * T_;

  // Q fragments for all 4 dk-slabs stay resident in VGPRs.
  V16 qf[4];
  {
    const __bf16* qrow = qb + (size_t)(qrow0 + lm) * DK_;
    #pragma unroll
    for (int s = 0; s < 4; ++s) {
      qf[s].h[0] = *(const v8bf*)(qrow + s * 32 + hi * 8);
      qf[s].h[1] = *(const v8bf*)(qrow + s * 32 + 16 + hi * 8);
    }
  }

  v8f o[8];
  #pragma unroll
  for (int f = 0; f < 8; ++f) o[f] = (v8f){};
  float mrow[8], lrow[8];
  #pragma unroll
  for (int r = 0; r < 8; ++r) { mrow[r] = -1e30f; lrow[r] = 0.0f; }

  const float scale = 0.08838834764831845f;    // 1/sqrt(128)

  for (int kv0 = 0; kv0 <= qrow0 + 15; kv0 += 64) {
    // ---- S = Q @ K^T : 4 column-fragments of 16 ----
    v8f sfr[4];
    #pragma unroll
    for (int f = 0; f < 4; ++f) sfr[f] = (v8f){};
    #pragma unroll
    for (int f = 0; f < 4; ++f) {
      const __bf16* krow = kb + (size_t)(kv0 + f * 16 + lm) * DK_;
      #pragma unroll
      for (int s = 0; s < 4; ++s) {
        V16 bfrag;
        bfrag.h[0] = *(const v8bf*)(krow + s * 32 + hi * 8);
        bfrag.h[1] = *(const v8bf*)(krow + s * 32 + 16 + hi * 8);
        sfr[f] = wmma_bf16(qf[s], bfrag, sfr[f]);
      }
    }
    // prefetch next KV block of K into cache
    if (kv0 + 64 <= qrow0 + 15)
      __builtin_prefetch(kb + (size_t)(kv0 + 64 + lm) * DK_, 0, 3);

    // ---- scale + causal mask ----
    const bool needmask = (kv0 + 63 > qrow0);
    #pragma unroll
    for (int f = 0; f < 4; ++f)
      #pragma unroll
      for (int r = 0; r < 8; ++r) {
        float v = sfr[f][r] * scale;
        if (needmask) {
          const int qr = qrow0 + r + hi * 8;
          const int kc = kv0 + f * 16 + lm;
          if (kc > qr) v = -1e30f;
        }
        sfr[f][r] = v;
      }

    // ---- per-row block max (reduce across the 16 lanes of each half) ----
    float alpha[8];
    #pragma unroll
    for (int r = 0; r < 8; ++r) {
      float bm = sfr[0][r];
      #pragma unroll
      for (int f = 1; f < 4; ++f) bm = fmaxf(bm, sfr[f][r]);
      #pragma unroll
      for (int m = 1; m < 16; m <<= 1) bm = fmaxf(bm, __shfl_xor(bm, m, 32));
      const float mnew = fmaxf(mrow[r], bm);
      alpha[r] = __expf(mrow[r] - mnew);
      mrow[r]  = mnew;
    }

    // ---- P = exp(S - m): stage to LDS (C-layout -> A-layout), row sums ----
    float rsum[8];
    #pragma unroll
    for (int r = 0; r < 8; ++r) rsum[r] = 0.0f;
    #pragma unroll
    for (int f = 0; f < 4; ++f)
      #pragma unroll
      for (int r = 0; r < 8; ++r) {
        const float p = __expf(sfr[f][r] - mrow[r]);
        rsum[r] += p;
        p_lds[wave][r + hi * 8][f * 16 + lm] = (__bf16)p;
      }
    #pragma unroll
    for (int r = 0; r < 8; ++r) {
      float s = rsum[r];
      #pragma unroll
      for (int m = 1; m < 16; m <<= 1) s += __shfl_xor(s, m, 32);
      lrow[r] = lrow[r] * alpha[r] + s;
    }

    // ---- rescale O accumulators ----
    #pragma unroll
    for (int f = 0; f < 8; ++f)
      #pragma unroll
      for (int r = 0; r < 8; ++r) o[f][r] *= alpha[r];

    // ---- reload P as A-fragments (wave-private LDS, no barrier needed) ----
    V16 pa[2];
    {
      const __bf16* prow = &p_lds[wave][lm][0];
      #pragma unroll
      for (int s = 0; s < 2; ++s) {
        pa[s].h[0] = *(const v8bf*)(prow + s * 32 + hi * 8);
        pa[s].h[1] = *(const v8bf*)(prow + s * 32 + 16 + hi * 8);
      }
    }

    // ---- O += P @ V : V^T rows are contiguous in the s dimension ----
    #pragma unroll
    for (int f = 0; f < 8; ++f) {
      const __bf16* vrow = vb + (size_t)(f * 16 + lm) * T_ + kv0;
      #pragma unroll
      for (int s = 0; s < 2; ++s) {
        V16 bfrag;
        bfrag.h[0] = *(const v8bf*)(vrow + s * 32 + hi * 8);
        bfrag.h[1] = *(const v8bf*)(vrow + s * 32 + 16 + hi * 8);
        o[f] = wmma_bf16(pa[s], bfrag, o[f]);
      }
    }
  }

  // ---- epilogue: normalize, restage via wave-private LDS, store float4 ----
  #pragma unroll
  for (int r = 0; r < 8; ++r) {
    const float inv_l = 1.0f / lrow[r];
    #pragma unroll
    for (int f = 0; f < 8; ++f)
      o_lds[wave][r + hi * 8][f * 16 + lm] = o[f][r] * inv_l;
  }
  float* __restrict__ ob = out + (size_t)bb * T_ * DK_;
  #pragma unroll
  for (int i = 0; i < 16; ++i) {
    const float4 val = *(const float4*)&o_lds[wave][i][lane * 4];
    *(float4*)(ob + (size_t)(qrow0 + i) * DK_ + lane * 4) = val;
  }
}

// ---------------------------------------------------------------------------
extern "C" void kernel_launch(void* const* d_in, const int* in_sizes, int n_in,
                              void* d_out, int out_size, void* d_ws, size_t ws_size,
                              hipStream_t stream) {
  const float* x  = (const float*)d_in[0];   // [B, T, DM]
  const float* Wq = (const float*)d_in[1];   // [DM, DK]
  const float* Wk = (const float*)d_in[2];
  const float* Wv = (const float*)d_in[3];
  float* out      = (float*)d_out;           // [B, T, DK]

  // Workspace: Q, K (row-major [B*T, DK]) and V^T ([B][DK][T]) in bf16.
  const size_t n_qkv = (size_t)B_ * T_ * DK_;      // 4M elements each
  __bf16* qws  = (__bf16*)d_ws;
  __bf16* kws  = qws + n_qkv;
  __bf16* vtws = kws + n_qkv;                      // total 24 MB

  dim3 g1((B_ * T_) / 128, 2);
  proj_rowmajor_kernel<<<g1, 256, 0, stream>>>(x, Wq, Wk, qws, kws);
  proj_transpose_kernel<<<(B_ * T_) / 128, 256, 0, stream>>>(x, Wv, vtws);

  dim3 g2(T_ / 64, B_);
  attn_fwd_kernel<<<g2, 128, 0, stream>>>(qws, kws, vtws, out);
}
